// CausalMHSAWithKVCache_50672024158655
// MI455X (gfx1250) — matrix operand order repaired and checked by reading, not compiled, not run on hardware
//
#include <hip/hip_runtime.h>
#include <hip/hip_bf16.h>
#include <math.h>

typedef __attribute__((ext_vector_type(16))) __bf16 v16bf;
typedef __attribute__((ext_vector_type(8)))  __bf16 v8bf;
typedef __attribute__((ext_vector_type(8)))  float  v8f;

#define WMMA_BF16(a,b,c) \
  __builtin_amdgcn_wmma_f32_16x16x32_bf16(false,(a),false,(b),(short)0,(c),false,false)

// Problem constants (fixed by the reference)
static constexpr int Bb = 2, T = 2048, Dm = 2048, H = 16, Dh = 128;

__device__ inline v8f vzero8f() {
  v8f z = {0.f,0.f,0.f,0.f,0.f,0.f,0.f,0.f};
  return z;
}

// ---------------------------------------------------------------------------
// f32 -> bf16 downconvert
// ---------------------------------------------------------------------------
__global__ void cvt_f32_bf16(const float* __restrict__ s, __bf16* __restrict__ d, int n) {
  int i = blockIdx.x * blockDim.x + threadIdx.x;
  if (i < n) d[i] = (__bf16)s[i];
}

// ---------------------------------------------------------------------------
// QKV GEMM: qkv[r, c] = sum_k X[r,k] * Wqkv[c,k];  r in [0,4096), c in [0,6144)
// Register-blocked: one wave computes a 32x64 tile (2 M-subtiles x 4 N-subtiles),
// 8 WMMAs per K-step of 32 with A reused 4x and each B reused 2x.
// Epilogue scatters Q,K as [B,H,T,Dh] and V *transposed* as [B,H,Dh,T] so the
// flash-attention P@V operand becomes a contiguous vector load.
// ---------------------------------------------------------------------------
__global__ void qkv_gemm_wmma(const __bf16* __restrict__ Xb, const __bf16* __restrict__ Wb,
                              __bf16* __restrict__ Qb, __bf16* __restrict__ Kb,
                              __bf16* __restrict__ Vtb) {
  const int lane = threadIdx.x & 31;
  const int wv   = threadIdx.x >> 5;
  const int tile = blockIdx.x * 8 + wv;      // 128*96 = 12288 macro-tiles
  const int mt   = tile & 127;               // 128 M blocks of 32 rows
  const int nt   = tile >> 7;                // 96 N blocks of 64 cols
  const int row0 = mt * 32, col0 = nt * 64;
  const int  l16 = lane & 15;
  const bool hiL = lane >= 16;
  const int  kb0 = hiL ? 16 : 0;             // A layout K sub-base (contiguous)
  const int  kb1 = hiL ? 16 : 0;             // B layout K sub-base

  const __bf16* ap0 = Xb + (size_t)(row0 + l16) * Dm;       // A rows, K-contiguous
  const __bf16* ap1 = ap0 + (size_t)16 * Dm;
  const __bf16* bp[4];
  #pragma unroll
  for (int nn = 0; nn < 4; nn++)
    bp[nn] = Wb + (size_t)(col0 + nn * 16 + l16) * Dm;      // B cols (W is [N,K])

  v8f acc[2][4] = {};
  #pragma unroll 2
  for (int k0 = 0; k0 < Dm; k0 += 32) {
    __builtin_prefetch(ap0 + k0 + 512, 0, 1);
    __builtin_prefetch(bp[0] + k0 + 512, 0, 1);
    v8bf a0lo = *(const v8bf*)(ap0 + k0 + kb0);
    v8bf a0hi = *(const v8bf*)(ap0 + k0 + kb0 + 8);
    v8bf a1lo = *(const v8bf*)(ap1 + k0 + kb0);
    v8bf a1hi = *(const v8bf*)(ap1 + k0 + kb0 + 8);
    v16bf a0, a1;
    #pragma unroll
    for (int i = 0; i < 8; i++) {
      a0[i] = a0lo[i]; a0[i + 8] = a0hi[i];
      a1[i] = a1lo[i]; a1[i + 8] = a1hi[i];
    }
    #pragma unroll
    for (int nn = 0; nn < 4; nn++) {
      v16bf b = *(const v16bf*)(bp[nn] + k0 + kb1);
      acc[0][nn] = WMMA_BF16(a0, b, acc[0][nn]);
      acc[1][nn] = WMMA_BF16(a1, b, acc[1][nn]);
    }
  }

  #pragma unroll
  for (int mm = 0; mm < 2; mm++) {
    #pragma unroll
    for (int nn = 0; nn < 4; nn++) {
      #pragma unroll
      for (int j = 0; j < 8; j++) {
        int r    = row0 + mm * 16 + j + (hiL ? 8 : 0);
        int c    = col0 + nn * 16 + l16;
        int comp = c >> 11;              // 0:q 1:k 2:v
        int h    = (c >> 7) & 15;
        int dh   = c & 127;
        int bi   = r >> 11;
        int t    = r & 2047;
        size_t bh = (size_t)(bi * H + h);
        __bf16 v = (__bf16)acc[mm][nn][j];
        if (comp == 0)      Qb [(bh * T + t) * Dh + dh] = v;
        else if (comp == 1) Kb [(bh * T + t) * Dh + dh] = v;
        else                Vtb[(bh * Dh + dh) * T + t] = v;   // transposed V
      }
    }
  }
}

// ---------------------------------------------------------------------------
// Flash attention: one wave owns a 16-query tile of one (b,h); 32 keys/iter.
// QK^T: 8 WMMAs over Dh chunks; PV: 8 WMMAs over Dh chunks, with V read from
// the transposed Vt layout so B-operands are aligned 32B vector loads.
// P goes C-layout -> A-layout through a 1KB/wave LDS staging tile.
// ---------------------------------------------------------------------------
__global__ void flash_attn_wmma(const __bf16* __restrict__ Qb, const __bf16* __restrict__ Kb,
                                const __bf16* __restrict__ Vtb, __bf16* __restrict__ Ob) {
  __shared__ __align__(16) __bf16 plds[8][16 * 32];   // per-wave P tile

  const int lane = threadIdx.x & 31;
  const int wv   = threadIdx.x >> 5;
  const int w    = blockIdx.x * 8 + wv;   // 4096 waves
  const int qt   = w & 127;               // 128 query tiles per (b,h)
  const int bh   = w >> 7;                // 32 (b,h) pairs
  const int q0   = qt * 16;
  const int  l16 = lane & 15;
  const bool hiL = lane >= 16;
  const size_t base = (size_t)bh * T * Dh;        // Q/K base  [bh][t][dh]
  const __bf16* vtb = Vtb + (size_t)bh * Dh * T;  // Vt base   [bh][dh][t]
  const float SC = 0.08838834764831845f * 1.44269504088896340f;  // 1/sqrt(128)*log2(e)

  // Q tile in WMMA A-layout: 4 chunks of K=32 spanning Dh=128
  v16bf qa[4];
  {
    const __bf16* qp = Qb + base + (size_t)(q0 + l16) * Dh;
    const int kb0 = hiL ? 16 : 0;
    #pragma unroll
    for (int c = 0; c < 4; c++) {
      v8bf lo  = *(const v8bf*)(qp + 32 * c + kb0);
      v8bf hi2 = *(const v8bf*)(qp + 32 * c + kb0 + 8);
      #pragma unroll
      for (int i = 0; i < 8; i++) { qa[c][i] = lo[i]; qa[c][i + 8] = hi2[i]; }
    }
  }

  v8f acc[8] = {};
  float m[8], l[8];
  #pragma unroll
  for (int j = 0; j < 8; j++) { m[j] = -INFINITY; l[j] = 0.f; }

  const int kb1 = hiL ? 16 : 0;
  const int nk  = q0 + 16;                // causal bound: keys <= q0+15
  for (int kt = 0; kt < nk; kt += 32) {
    // ---- S = Q @ K^T for 32 keys (two 16-key halves) ----
    v8f S0 = vzero8f(), S1 = vzero8f();
    const __bf16* kp = Kb + base + (size_t)(kt + l16) * Dh + kb1;
    #pragma unroll
    for (int c = 0; c < 4; c++) {
      v16bf bk = *(const v16bf*)(kp + 32 * c);
      S0 = WMMA_BF16(qa[c], bk, S0);
    }
    #pragma unroll
    for (int c = 0; c < 4; c++) {
      v16bf bk = *(const v16bf*)(kp + (size_t)16 * Dh + 32 * c);
      S1 = WMMA_BF16(qa[c], bk, S1);
    }

    // ---- online softmax (row = VGPR index + lane-half) ----
    #pragma unroll
    for (int j = 0; j < 8; j++) {
      int rj   = j + (hiL ? 8 : 0);
      int qrow = q0 + rj;
      int key0 = kt + l16;
      float s0 = (key0      <= qrow) ? S0[j] * SC : -INFINITY;
      float s1 = (key0 + 16 <= qrow) ? S1[j] * SC : -INFINITY;
      float mx = fmaxf(s0, s1);
      mx = fmaxf(mx, __shfl_xor(mx, 1, 32));
      mx = fmaxf(mx, __shfl_xor(mx, 2, 32));
      mx = fmaxf(mx, __shfl_xor(mx, 4, 32));
      mx = fmaxf(mx, __shfl_xor(mx, 8, 32));
      float mn    = fmaxf(m[j], mx);
      float alpha = exp2f(m[j] - mn);
      float p0    = exp2f(s0 - mn);
      float p1    = exp2f(s1 - mn);
      float rs = p0 + p1;
      rs += __shfl_xor(rs, 1, 32);
      rs += __shfl_xor(rs, 2, 32);
      rs += __shfl_xor(rs, 4, 32);
      rs += __shfl_xor(rs, 8, 32);
      l[j] = l[j] * alpha + rs;
      m[j] = mn;
      plds[wv][rj * 32 + l16]      = (__bf16)p0;
      plds[wv][rj * 32 + 16 + l16] = (__bf16)p1;
      #pragma unroll
      for (int cc = 0; cc < 8; cc++) acc[cc][j] *= alpha;
    }

    // ---- reload P in A-layout (16x32, K = 32 keys) ----
    v16bf pa;
    {
      const int akb = hiL ? 16 : 0;
      v8bf lo  = *(const v8bf*)&plds[wv][l16 * 32 + akb];
      v8bf hi2 = *(const v8bf*)&plds[wv][l16 * 32 + akb + 8];
      #pragma unroll
      for (int i = 0; i < 8; i++) { pa[i] = lo[i]; pa[i + 8] = hi2[i]; }
    }

    // ---- acc += P @ V over 8 Dh chunks; Vt gives contiguous key runs ----
    #pragma unroll
    for (int cc = 0; cc < 8; cc++) {
      v16bf bv = *(const v16bf*)(vtb + (size_t)(cc * 16 + l16) * T + kt + kb1);
      acc[cc] = WMMA_BF16(pa, bv, acc[cc]);
    }
  }

  // ---- epilogue: normalize, store [B,T,D] with d = h*128 + dh ----
  const int b = bh >> 4, h = bh & 15;
  #pragma unroll
  for (int j = 0; j < 8; j++) {
    int rj = j + (hiL ? 8 : 0);
    size_t row = (size_t)b * T + q0 + rj;
    float inv = 1.0f / l[j];
    #pragma unroll
    for (int cc = 0; cc < 8; cc++) {
      Ob[row * Dm + h * Dh + cc * 16 + l16] = (__bf16)(acc[cc][j] * inv);
    }
  }
}

// ---------------------------------------------------------------------------
// Output projection: y[r,c] = sum_k O[r,k] * Wproj[c,k]; f32 output.
// Same 32x64 register blocking as the QKV GEMM.
// ---------------------------------------------------------------------------
__global__ void proj_gemm_wmma(const __bf16* __restrict__ Ob, const __bf16* __restrict__ Wb,
                               float* __restrict__ out) {
  const int lane = threadIdx.x & 31;
  const int wv   = threadIdx.x >> 5;
  const int tile = blockIdx.x * 8 + wv;     // 128*32 = 4096 macro-tiles
  const int mt   = tile & 127;
  const int nt   = tile >> 7;               // 32 N blocks of 64
  const int row0 = mt * 32, col0 = nt * 64;
  const int  l16 = lane & 15;
  const bool hiL = lane >= 16;
  const int  kb0 = hiL ? 16 : 0;
  const int  kb1 = hiL ? 16 : 0;

  const __bf16* ap0 = Ob + (size_t)(row0 + l16) * Dm;
  const __bf16* ap1 = ap0 + (size_t)16 * Dm;
  const __bf16* bp[4];
  #pragma unroll
  for (int nn = 0; nn < 4; nn++)
    bp[nn] = Wb + (size_t)(col0 + nn * 16 + l16) * Dm;

  v8f acc[2][4] = {};
  #pragma unroll 2
  for (int k0 = 0; k0 < Dm; k0 += 32) {
    __builtin_prefetch(ap0 + k0 + 512, 0, 1);
    __builtin_prefetch(bp[0] + k0 + 512, 0, 1);
    v8bf a0lo = *(const v8bf*)(ap0 + k0 + kb0);
    v8bf a0hi = *(const v8bf*)(ap0 + k0 + kb0 + 8);
    v8bf a1lo = *(const v8bf*)(ap1 + k0 + kb0);
    v8bf a1hi = *(const v8bf*)(ap1 + k0 + kb0 + 8);
    v16bf a0, a1;
    #pragma unroll
    for (int i = 0; i < 8; i++) {
      a0[i] = a0lo[i]; a0[i + 8] = a0hi[i];
      a1[i] = a1lo[i]; a1[i + 8] = a1hi[i];
    }
    #pragma unroll
    for (int nn = 0; nn < 4; nn++) {
      v16bf b = *(const v16bf*)(bp[nn] + k0 + kb1);
      acc[0][nn] = WMMA_BF16(a0, b, acc[0][nn]);
      acc[1][nn] = WMMA_BF16(a1, b, acc[1][nn]);
    }
  }

  #pragma unroll
  for (int mm = 0; mm < 2; mm++) {
    #pragma unroll
    for (int nn = 0; nn < 4; nn++) {
      #pragma unroll
      for (int j = 0; j < 8; j++) {
        int r = row0 + mm * 16 + j + (hiL ? 8 : 0);
        int c = col0 + nn * 16 + l16;
        out[(size_t)r * Dm + c] = acc[mm][nn][j];
      }
    }
  }
}

// ---------------------------------------------------------------------------
// Launcher
// ---------------------------------------------------------------------------
extern "C" void kernel_launch(void* const* d_in, const int* in_sizes, int n_in,
                              void* d_out, int out_size, void* d_ws, size_t ws_size,
                              hipStream_t stream) {
  const float* x     = (const float*)d_in[0];   // [2,2048,2048]
  const float* Wqkv  = (const float*)d_in[1];   // [6144,2048]
  const float* Wproj = (const float*)d_in[2];   // [2048,2048]
  float* out = (float*)d_out;                   // [2,2048,2048] f32

  // workspace layout (bytes):
  //   Xb     @ 0          : 4096*2048  bf16 = 16.78 MB
  //   Wqkvb  @ 16777216   : 6144*2048  bf16 = 25.17 MB
  //   Wprojb @ 41943040   : 2048*2048  bf16 =  8.39 MB
  //   Q,K,Vt @ 50331648   : 3 * [2,16,2048,128] bf16 = 50.33 MB
  //   Ob     @ 100663296  : 4096*2048  bf16 = 16.78 MB   (total 112 MB)
  char* w = (char*)d_ws;
  __bf16* Xb  = (__bf16*)(w);
  __bf16* Wqb = (__bf16*)(w + 16777216);
  __bf16* Wpb = (__bf16*)(w + 41943040);
  __bf16* Qb  = (__bf16*)(w + 50331648);
  __bf16* Kb  = Qb + (size_t)2 * 16 * 2048 * 128;
  __bf16* Vtb = Kb + (size_t)2 * 16 * 2048 * 128;
  __bf16* Ob  = (__bf16*)(w + 100663296);

  cvt_f32_bf16<<<8388608 / 256, 256, 0, stream>>>(x, Xb, 8388608);
  cvt_f32_bf16<<<12582912 / 256, 256, 0, stream>>>(Wqkv, Wqb, 12582912);
  cvt_f32_bf16<<<4194304 / 256, 256, 0, stream>>>(Wproj, Wpb, 4194304);

  // 128 * 96 = 12288 macro-tiles (32x64), 8 waves/block
  qkv_gemm_wmma<<<1536, 256, 0, stream>>>(Xb, Wqb, Qb, Kb, Vtb);

  // 32 (b,h) * 128 q-tiles = 4096 waves, 8 waves/block
  flash_attn_wmma<<<512, 256, 0, stream>>>(Qb, Kb, Vtb, Ob);

  // 128 * 32 = 4096 macro-tiles (32x64), 8 waves/block
  proj_gemm_wmma<<<512, 256, 0, stream>>>(Ob, Wpb, out);
}
